// FlashAttention_19756849562332
// MI455X (gfx1250) — matrix operand order repaired and checked by reading, not compiled, not run on hardware
//
#include <hip/hip_runtime.h>
#include <hip/hip_bf16.h>

typedef __attribute__((ext_vector_type(16))) _Float16 v16h;
typedef __attribute__((ext_vector_type(8)))  _Float16 v8h;
typedef __attribute__((ext_vector_type(8)))  float    v8f;
typedef __attribute__((ext_vector_type(4)))  unsigned int u32x4;
typedef __attribute__((ext_vector_type(8)))  int      i32x8;
typedef __attribute__((ext_vector_type(4)))  int      i32x4;

#define BATCH 8
#define SEQ 1024
#define DMODEL 1024
#define NHEADS 16
#define HDIM 64

#if __has_builtin(__builtin_amdgcn_tensor_load_to_lds)
#define USE_TDM 1
#else
#define USE_TDM 0
#endif

// ---- WMMA 16-bit fragment layouts used by this code (wave32; contiguous k, not AMD's table order) ----
// A 16x32 f16: lane L row M=L%16, group g=L/16; half h -> K = g*16+h
// B 32x16 f16: lane L col N=L%16; half h -> K = g*16+h  => one contiguous 16-half run.
// C/D f32 16x16: reg r -> row M = r+8g, col N = lane%16.

__device__ __forceinline__ v16h load_a_frag(const _Float16* row, int g) {
    v8h lo = *(const v8h*)(row + g * 16);
    v8h hi = *(const v8h*)(row + g * 16 + 8);
    return __builtin_shufflevector(lo, hi, 0, 1, 2, 3, 4, 5, 6, 7,
                                   8, 9, 10, 11, 12, 13, 14, 15);
}

// ---- Tensor Data Mover: 2D f16 tile (tile_d0 contiguous elems x tile_d1 rows)
// from global (row stride = stride_elems) into LDS at lds_byte_addr. ----
__device__ __forceinline__ void tdm_load_2d_f16(unsigned int lds_byte_addr,
                                                const _Float16* gaddr,
                                                unsigned int tile_d0,
                                                unsigned int tile_d1,
                                                unsigned int stride_elems) {
#if USE_TDM
    unsigned long long ga = (unsigned long long)(uintptr_t)gaddr;
    u32x4 g0;
    g0[0] = 1u;                                            // count=1, user D#
    g0[1] = lds_byte_addr;                                 // D#.lds_addr
    g0[2] = (unsigned int)ga;                              // global_addr[31:0]
    g0[3] = ((unsigned int)(ga >> 32) & 0x01FFFFFFu) | (2u << 30); // addr hi | type=2
    i32x8 g1;
    g1[0] = (int)(1u << 16);                               // data_size = 2 bytes
    g1[1] = (int)((tile_d0 & 0xFFFFu) << 16);              // tensor_dim0[15:0]
    g1[2] = (int)((tile_d0 >> 16) | ((tile_d1 & 0xFFFFu) << 16)); // dim0 hi | dim1 lo
    g1[3] = (int)((tile_d1 >> 16) | ((tile_d0 & 0xFFFFu) << 16)); // dim1 hi | tile_dim0
    g1[4] = (int)(tile_d1 & 0xFFFFu);                      // tile_dim1 (tile_dim2=0)
    g1[5] = (int)stride_elems;                             // tensor_dim0_stride lo
    g1[6] = 0;
    g1[7] = 0;
    i32x4 zz = (i32x4)0;
#if __clang_major__ >= 23
    i32x8 z8 = (i32x8)0;
    __builtin_amdgcn_tensor_load_to_lds(g0, g1, zz, zz, z8, 0);
#else
    __builtin_amdgcn_tensor_load_to_lds(g0, g1, zz, zz, 0);
#endif
#endif
}

// Wait until the OLDEST in-flight tile-pair has landed: with one tile-pair
// (2 TDM ops) still in flight behind it, waiting for cnt<=2 drains the older
// pair (TENSORcnt completes in order). Last iteration waits for 0.
__device__ __forceinline__ void tdm_wait_pair(bool more_in_flight) {
#if USE_TDM && __has_builtin(__builtin_amdgcn_s_wait_tensorcnt)
    if (more_in_flight) __builtin_amdgcn_s_wait_tensorcnt(2);
    else                __builtin_amdgcn_s_wait_tensorcnt(0);
#endif
}

// ---------------- fp32 -> f16 conversion ----------------
__global__ void cvt_f32_to_f16(const float* __restrict__ in,
                               _Float16* __restrict__ out, int n) {
    int i = blockIdx.x * blockDim.x + threadIdx.x;
    if (i < n) out[i] = (_Float16)in[i];
}

// ---- fused fp32->f16 + transpose for weights: in[K][N] -> out[N][K] ----
__global__ void cvt_transpose_w(const float* __restrict__ in,
                                _Float16* __restrict__ out, int K, int N) {
    __shared__ _Float16 t[64][65];
    const int tid = threadIdx.x;
    const int kB = blockIdx.y * 64, nB = blockIdx.x * 64;
    for (int i = 0; i < 32; ++i) {
        int q = tid + i * 128, r = q >> 6, c = q & 63;
        t[r][c] = (_Float16)in[(size_t)(kB + r) * N + nB + c];
    }
    __syncthreads();
    for (int i = 0; i < 32; ++i) {
        int q = tid + i * 128, r = q >> 6, c = q & 63;   // r: N-row, c: K-col
        out[(size_t)(nB + r) * K + kB + c] = t[c][r];
    }
}

// ---- build Vt[bh][dim][seq] from V[B,S,H*HDIM] ----
__global__ void build_vt(const _Float16* __restrict__ V,
                         _Float16* __restrict__ Vt) {
    __shared__ _Float16 t[64][65];
    const int tid = threadIdx.x;
    const int bh = blockIdx.y;
    const int b = bh / NHEADS, h = bh % NHEADS;
    const int s0 = blockIdx.x * 64;
    for (int i = 0; i < 32; ++i) {
        int q = tid + i * 128, si = q >> 6, d = q & 63;
        t[si][d] = V[((size_t)b * SEQ + s0 + si) * DMODEL + h * HDIM + d];
    }
    __syncthreads();
    for (int i = 0; i < 32; ++i) {
        int q = tid + i * 128, d = q >> 6, ss = q & 63;
        Vt[((size_t)bh * HDIM + d) * SEQ + s0 + ss] = t[ss][d];
    }
}

// ---------------- WMMA GEMM: C[M,N] = A[M,K] @ Bt[N,K]^T + bias ----------------
// block = 128 threads (4 waves). Block tile 128(M) x 64(N); wave tile 32 x 64.
// grid = (N/64, M/128). Bt is pre-transposed. Double-buffered TDM staging.
__global__ void gemm_f16_wmma(const _Float16* __restrict__ A,
                              const _Float16* __restrict__ Bt,
                              const float* __restrict__ bias,
                              _Float16* __restrict__ Ch,   // f16 output (or null)
                              float* __restrict__ Cf,      // f32 output (or null)
                              int M, int N, int K) {
    __shared__ _Float16 lA[2][128 * 32];   // 16 KB
    __shared__ _Float16 lBt[2][64 * 32];   //  8 KB

    const int tid  = threadIdx.x;
    const int lane = tid & 31;
    const int w    = tid >> 5;
    const int m    = lane & 15;
    const int g    = lane >> 4;
    const int n    = lane & 15;

    const int mBlock = blockIdx.y * 128;
    const int nBlock = blockIdx.x * 64;

    v8f acc[2][4];
    for (int s = 0; s < 2; ++s)
        for (int t = 0; t < 4; ++t) acc[s][t] = (v8f)0.0f;

    unsigned int lA_addr[2], lBt_addr[2];
    for (int p = 0; p < 2; ++p) {
        lA_addr[p]  = (unsigned int)(uintptr_t)&lA[p][0];
        lBt_addr[p] = (unsigned int)(uintptr_t)&lBt[p][0];
    }

#if USE_TDM
    if (w == 0) {   // prologue: tile 0 -> buffer 0
        tdm_load_2d_f16(lA_addr[0],  A  + (size_t)mBlock * K, 32, 128, K);
        tdm_load_2d_f16(lBt_addr[0], Bt + (size_t)nBlock * K, 32, 64,  K);
    }
#endif

    for (int k0 = 0; k0 < K; k0 += 32) {
        const int buf = (k0 >> 5) & 1;
#if USE_TDM
        const bool more = (k0 + 32) < K;
        __syncthreads();   // everyone done reading buf^1 (overwritten next)
        if (w == 0) {
            if (more) {
                tdm_load_2d_f16(lA_addr[buf ^ 1],
                                A + (size_t)mBlock * K + k0 + 32, 32, 128, K);
                tdm_load_2d_f16(lBt_addr[buf ^ 1],
                                Bt + (size_t)nBlock * K + k0 + 32, 32, 64, K);
            }
            tdm_wait_pair(more);   // tile k0 has landed
        }
        __syncthreads();   // publish tile k0
#else
        uint32_t* lAu  = (uint32_t*)&lA[buf][0];
        uint32_t* lBtu = (uint32_t*)&lBt[buf][0];
        for (int i = 0; i < 16; ++i) {
            int q = tid + i * 128, ar = q >> 4, ac = q & 15;
            lAu[q] = ((const uint32_t*)(A + (size_t)(mBlock + ar) * K + k0))[ac];
        }
        for (int i = 0; i < 8; ++i) {
            int q = tid + i * 128, nr = q >> 4, nc = q & 15;
            lBtu[q] = ((const uint32_t*)(Bt + (size_t)(nBlock + nr) * K + k0))[nc];
        }
        __syncthreads();
#endif

        v16h a[2];
        for (int s = 0; s < 2; ++s)
            a[s] = load_a_frag(&lA[buf][0] + (w * 32 + s * 16 + m) * 32, g);

        for (int t = 0; t < 4; ++t) {
            v16h b = *(const v16h*)(&lBt[buf][0] + (t * 16 + n) * 32 + g * 16);
            for (int s = 0; s < 2; ++s)
                acc[s][t] = __builtin_amdgcn_wmma_f32_16x16x32_f16(
                    false, a[s], false, b, (short)0, acc[s][t], false, false);
        }
#if !USE_TDM
        __syncthreads();
#endif
    }

    for (int t = 0; t < 4; ++t) {
        int col = nBlock + t * 16 + n;
        float bv = bias[col];
        for (int s = 0; s < 2; ++s) {
            for (int r = 0; r < 8; ++r) {
                int row = mBlock + w * 32 + s * 16 + r + 8 * g;
                float v = acc[s][t][r] + bv;
                if (Ch) Ch[(size_t)row * N + col] = (_Float16)v;
                else    Cf[(size_t)row * N + col] = v;
            }
        }
    }
}

// ---------------- Flash attention ----------------
// Q,K in [B,S,H*HDIM] f16; Vt in [B*H][HDIM][SEQ] f16. grid (S/64, B*H), block 128.
__global__ void flash_attn_wmma(const _Float16* __restrict__ Q,
                                const _Float16* __restrict__ Km,
                                const _Float16* __restrict__ Vt,
                                _Float16* __restrict__ O) {
    __shared__ _Float16 lK[2][32 * 64];   // 8 KB
    __shared__ _Float16 lVt[2][64 * 32];  // 8 KB
    __shared__ _Float16 lP[4 * 16 * 32];  // 4 KB

    const int tid  = threadIdx.x;
    const int lane = tid & 31;
    const int w    = tid >> 5;
    const int m    = lane & 15;
    const int g    = lane >> 4;
    const int n    = lane & 15;

    const int bh = blockIdx.y;
    const int b  = bh / NHEADS;
    const int h  = bh % NHEADS;
    const int qbase = blockIdx.x * 64 + w * 16;

    const size_t base = (size_t)b * SEQ * DMODEL + (size_t)h * HDIM;
    const _Float16* Qp  = Q  + base;
    const _Float16* Kp  = Km + base;
    const _Float16* VtP = Vt + (size_t)bh * HDIM * SEQ;
    _Float16*       Op  = O  + base;

    // Q fragments (16x64 as two 16x32 A-frags), pre-scaled by 1/sqrt(HDIM)
    v16h aQ0, aQ1;
    {
        const _Float16* qrow = Qp + (size_t)(qbase + m) * DMODEL;
        aQ0 = load_a_frag(qrow, g);
        aQ1 = load_a_frag(qrow + 32, g);
        const _Float16 sc = (_Float16)0.125f;
        aQ0 *= sc;
        aQ1 *= sc;
    }

    v8f acc[4];
    for (int t = 0; t < 4; ++t) acc[t] = (v8f)0.0f;
    float mrow[8], lrow[8];
    for (int r = 0; r < 8; ++r) { mrow[r] = -3.0e38f; lrow[r] = 0.0f; }

    unsigned int lK_addr[2], lVt_addr[2];
    for (int p = 0; p < 2; ++p) {
        lK_addr[p]  = (unsigned int)(uintptr_t)&lK[p][0];
        lVt_addr[p] = (unsigned int)(uintptr_t)&lVt[p][0];
    }

#if USE_TDM
    if (w == 0) {   // prologue: key-block 0 -> buffer 0
        tdm_load_2d_f16(lK_addr[0],  Kp,  64, 32, DMODEL);
        tdm_load_2d_f16(lVt_addr[0], VtP, 32, 64, SEQ);
    }
#endif

    for (int kb = 0; kb < SEQ; kb += 32) {
        const int buf = (kb >> 5) & 1;
#if USE_TDM
        const bool more = (kb + 32) < SEQ;
        __syncthreads();
        if (w == 0) {
            if (more) {
                tdm_load_2d_f16(lK_addr[buf ^ 1],
                                Kp + (size_t)(kb + 32) * DMODEL, 64, 32, DMODEL);
                tdm_load_2d_f16(lVt_addr[buf ^ 1],
                                VtP + kb + 32, 32, 64, SEQ);
            }
            tdm_wait_pair(more);
        }
        __syncthreads();
#else
        uint32_t* lKu  = (uint32_t*)&lK[buf][0];
        uint32_t* lVtu = (uint32_t*)&lVt[buf][0];
        for (int i = 0; i < 8; ++i) {
            int q = tid + i * 128, kr = q >> 5, kc = q & 31;
            lKu[q] = ((const uint32_t*)(Kp + (size_t)(kb + kr) * DMODEL))[kc];
        }
        for (int i = 0; i < 8; ++i) {
            int q = tid + i * 128, dr = q >> 4, dc = q & 15;
            lVtu[q] = ((const uint32_t*)(VtP + (size_t)dr * SEQ + kb))[dc];
        }
        __syncthreads();
#endif

        // ---- scores: two 16x16 tiles, K-dim 64 chained as two x32 WMMAs ----
        v8f s[2];
        for (int u = 0; u < 2; ++u) {
            const _Float16* krow = &lK[buf][0] + (u * 16 + n) * 64;
            v16h bLo = *(const v16h*)(krow + g * 16);
            v16h bHi = *(const v16h*)(krow + 32 + g * 16);
            v8f c = (v8f)0.0f;
            c = __builtin_amdgcn_wmma_f32_16x16x32_f16(false, aQ1, false, bHi,
                                                       (short)0, c, false, false);
            c = __builtin_amdgcn_wmma_f32_16x16x32_f16(false, aQ0, false, bLo,
                                                       (short)0, c, false, false);
            s[u] = c;
        }

        // ---- online softmax per row (row M = r + 8g) ----
        float alpha[8];
        for (int r = 0; r < 8; ++r) {
            float cur = fmaxf(s[0][r], s[1][r]);
            for (int off = 8; off >= 1; off >>= 1)
                cur = fmaxf(cur, __shfl_xor(cur, off, 16));
            float nm = fmaxf(mrow[r], cur);
            alpha[r] = __expf(mrow[r] - nm);
            mrow[r]  = nm;
            float p0 = __expf(s[0][r] - nm);
            float p1 = __expf(s[1][r] - nm);
            float rs = p0 + p1;
            for (int off = 8; off >= 1; off >>= 1)
                rs += __shfl_xor(rs, off, 16);
            lrow[r] = lrow[r] * alpha[r] + rs;
            lP[w * 512 + (r + 8 * g) * 32 + n]      = (_Float16)p0;
            lP[w * 512 + (r + 8 * g) * 32 + 16 + n] = (_Float16)p1;
        }
        for (int t = 0; t < 4; ++t)
            for (int r = 0; r < 8; ++r)
                acc[t][r] *= alpha[r];

        // ---- P (16x32) @ V (32x64) ----
        v16h aP = load_a_frag(lP + w * 512 + m * 32, g);
        for (int t = 0; t < 4; ++t) {
            v16h bV = *(const v16h*)(&lVt[buf][0] + (t * 16 + n) * 32 + g * 16);
            acc[t] = __builtin_amdgcn_wmma_f32_16x16x32_f16(
                false, aP, false, bV, (short)0, acc[t], false, false);
        }
#if !USE_TDM
        __syncthreads();
#endif
    }

    for (int t = 0; t < 4; ++t) {
        for (int r = 0; r < 8; ++r) {
            float o = acc[t][r] / lrow[r];
            Op[(size_t)(qbase + r + 8 * g) * DMODEL + t * 16 + n] = (_Float16)o;
        }
    }
}

// ---------------- host-side launcher ----------------
extern "C" void kernel_launch(void* const* d_in, const int* in_sizes, int n_in,
                              void* d_out, int out_size, void* d_ws, size_t ws_size,
                              hipStream_t stream) {
    const float* X  = (const float*)d_in[0];
    const float* Wq = (const float*)d_in[1];
    const float* bq = (const float*)d_in[2];
    const float* Wk = (const float*)d_in[3];
    const float* bk = (const float*)d_in[4];
    const float* Wv = (const float*)d_in[5];
    const float* bv = (const float*)d_in[6];
    const float* Wo = (const float*)d_in[7];
    const float* bo = (const float*)d_in[8];
    float* out = (float*)d_out;

    const int M = BATCH * SEQ;         // 8192
    const int N = DMODEL;              // 1024
    const int K = DMODEL;              // 1024
    const size_t MD = (size_t)M * N;   // 8M elements
    const size_t WD = (size_t)N * K;   // 1M elements

    _Float16* ws = (_Float16*)d_ws;
    _Float16* Xh  = ws;            ws += MD;
    _Float16* WqT = ws;            ws += WD;
    _Float16* WkT = ws;            ws += WD;
    _Float16* WvT = ws;            ws += WD;
    _Float16* WoT = ws;            ws += WD;
    _Float16* Qh  = ws;            ws += MD;
    _Float16* Kh  = ws;            ws += MD;
    _Float16* Vh  = ws;            ws += MD;
    _Float16* Vtg = ws;            ws += MD;
    _Float16* Ah  = ws;            ws += MD;

    cvt_f32_to_f16<<<(int)((MD + 255) / 256), 256, 0, stream>>>(X, Xh, (int)MD);
    dim3 wgrid(N / 64, K / 64);
    cvt_transpose_w<<<wgrid, 128, 0, stream>>>(Wq, WqT, K, N);
    cvt_transpose_w<<<wgrid, 128, 0, stream>>>(Wk, WkT, K, N);
    cvt_transpose_w<<<wgrid, 128, 0, stream>>>(Wv, WvT, K, N);
    cvt_transpose_w<<<wgrid, 128, 0, stream>>>(Wo, WoT, K, N);

    dim3 ggrid(N / 64, M / 128);
    gemm_f16_wmma<<<ggrid, 128, 0, stream>>>(Xh, WqT, bq, Qh, nullptr, M, N, K);
    gemm_f16_wmma<<<ggrid, 128, 0, stream>>>(Xh, WkT, bk, Kh, nullptr, M, N, K);
    gemm_f16_wmma<<<ggrid, 128, 0, stream>>>(Xh, WvT, bv, Vh, nullptr, M, N, K);

    dim3 vgrid(SEQ / 64, BATCH * NHEADS);
    build_vt<<<vgrid, 128, 0, stream>>>(Vh, Vtg);

    dim3 agrid(SEQ / 64, BATCH * NHEADS);
    flash_attn_wmma<<<agrid, 128, 0, stream>>>(Qh, Kh, Vtg, Ah);

    gemm_f16_wmma<<<ggrid, 128, 0, stream>>>(Ah, WoT, bo, nullptr, out, M, N, K);
}
